// SoftDecisionTree_70746701299772
// MI455X (gfx1250) — matrix operand order repaired and checked by reading, not compiled, not run on hardware
//
#include <hip/hip_runtime.h>
#include <hip/hip_bf16.h>

typedef __attribute__((ext_vector_type(16))) _Float16 v16h;
typedef __attribute__((ext_vector_type(8)))  _Float16 v8h;
typedef __attribute__((ext_vector_type(4)))  _Float16 v4h;
typedef __attribute__((ext_vector_type(8)))  float    v8f;

#define BS  16384
#define DD  512
#define NI  255
#define NL  256
#define NC  1000
#define RED_BLOCKS 128
#define ROWS_PER_RED (BS / RED_BLOCKS)   // 128

// ---------------- Stage 1: column-wise partial sums (deterministic) ----------
__global__ __launch_bounds__(DD) void colstat_partial(const float* __restrict__ x,
                                                      float* __restrict__ psum,
                                                      float* __restrict__ psq) {
    const int c   = threadIdx.x;       // 0..511 (one column per thread)
    const int blk = blockIdx.x;        // 0..127 (row chunk)
    const float* xp = x + (size_t)blk * ROWS_PER_RED * DD + c;
    float s = 0.f, s2 = 0.f;
    #pragma unroll 4
    for (int r = 0; r < ROWS_PER_RED; ++r) {
        float v = xp[(size_t)r * DD];
        s += v; s2 += v * v;
    }
    psum[blk * DD + c] = s;
    psq [blk * DD + c] = s2;
}

// ---------------- Stage 2: finalize mean/var -> per-column scale/shift -------
__global__ __launch_bounds__(256) void colstat_final(const float* __restrict__ psum,
                                                     const float* __restrict__ psq,
                                                     const float* __restrict__ gamma,
                                                     const float* __restrict__ beta,
                                                     float* __restrict__ scale,
                                                     float* __restrict__ shift) {
    int c = blockIdx.x * blockDim.x + threadIdx.x;
    if (c >= DD) return;
    float s = 0.f, s2 = 0.f;
    for (int b = 0; b < RED_BLOCKS; ++b) { s += psum[b * DD + c]; s2 += psq[b * DD + c]; }
    float mean = s * (1.f / BS);
    float var  = s2 * (1.f / BS) - mean * mean;
    float sc   = gamma[c] * rsqrtf(var + 1e-5f);
    scale[c] = sc;
    shift[c] = beta[c] - mean * sc;
}

// ---------------- W (255x512 f32) -> padded f16 (256x512) --------------------
__global__ __launch_bounds__(256) void wconv(const float* __restrict__ W,
                                             _Float16* __restrict__ W16) {
    int e = blockIdx.x * blockDim.x + threadIdx.x;  // 0..131071
    int n = e >> 9, k = e & (DD - 1);
    float v = (n < NI) ? W[n * DD + k] : 0.f;
    W16[e] = (_Float16)v;
}

// ---------------- softmax table: 256 x 1000 ---------------------------------
__global__ __launch_bounds__(256) void softmax_rows(const float* __restrict__ cr,
                                                    float* __restrict__ dist) {
    __shared__ float red[256];
    const int l = blockIdx.x, t = threadIdx.x;
    const float* rp = cr + (size_t)l * NC;
    float m = -3.402823466e+38f;
    for (int c = t; c < NC; c += 256) m = fmaxf(m, rp[c]);
    red[t] = m; __syncthreads();
    for (int s = 128; s > 0; s >>= 1) { if (t < s) red[t] = fmaxf(red[t], red[t + s]); __syncthreads(); }
    m = red[0]; __syncthreads();
    float sum = 0.f;
    for (int c = t; c < NC; c += 256) sum += __expf(rp[c] - m);
    red[t] = sum; __syncthreads();
    for (int s = 128; s > 0; s >>= 1) { if (t < s) red[t] += red[t + s]; __syncthreads(); }
    float inv = 1.f / red[0];
    float* op = dist + (size_t)l * NC;
    for (int c = t; c < NC; c += 256) op[c] = __expf(rp[c] - m) * inv;
}

// ---------------- fused: normalize -> WMMA gates -> sigmoid -> leaf argmax ---
// block = 128 threads (4 waves), 16 rows per block, 256 gate columns.
__global__ __launch_bounds__(128) void tree_gates_argmax(
        const float* __restrict__ x, const _Float16* __restrict__ W16,
        const float* __restrict__ bias, const float* __restrict__ scale,
        const float* __restrict__ shift, int* __restrict__ leafidx) {
    __shared__ _Float16 lA[16 * DD];       // 16 KB: normalized f16 A tile
    __shared__ float    lgates[16 * NL];   // 16 KB: sigmoid gates

    const int tid  = threadIdx.x;
    const int row0 = blockIdx.x * 16;

    // ---- stage normalized x tile as f16 into LDS (float4 loads) ----
    for (int i = tid; i < 16 * DD / 4; i += 128) {
        const int flat = i * 4;
        const int r = flat >> 9, c = flat & (DD - 1);
        const float4 xv = *(const float4*)(x + (size_t)(row0 + r) * DD + c);
        const float4 sv = *(const float4*)(scale + c);
        const float4 tv = *(const float4*)(shift + c);
        v4h h;
        h[0] = (_Float16)(xv.x * sv.x + tv.x);
        h[1] = (_Float16)(xv.y * sv.y + tv.y);
        h[2] = (_Float16)(xv.z * sv.z + tv.z);
        h[3] = (_Float16)(xv.w * sv.w + tv.w);
        *(v4h*)(&lA[flat]) = h;
    }
    __syncthreads();

    const int lane  = tid & 31;
    const int wv    = tid >> 5;
    const int nbase = wv * 64;            // each wave: 4 N-tiles = 64 cols
    const int mA    = lane & 15;
    const int koffA = (lane >> 4) * 8;    // A frag: K in {koffA..+7, koffA+16..+23}
    const int nB    = lane & 15;
    const int koffB = (lane >> 4) * 8;    // B frag: K in {koffB..+7, koffB+16..+23}

    v8f acc[4] = {v8f{}, v8f{}, v8f{}, v8f{}};

    for (int k0 = 0; k0 < DD; k0 += 32) {
        union { v16h v; v8h h[2]; } a;
        const _Float16* ap = &lA[mA * DD + k0 + koffA];
        a.h[0] = *(const v8h*)(ap);
        a.h[1] = *(const v8h*)(ap + 16);
        #pragma unroll
        for (int j = 0; j < 4; ++j) {
            union { v16h v; v8h h[2]; } bf;
            const _Float16* bp = W16 + (size_t)(nbase + j * 16 + nB) * DD + k0 + koffB;
            bf.h[0] = *(const v8h*)(bp);
            bf.h[1] = *(const v8h*)(bp + 16);
            acc[j] = __builtin_amdgcn_wmma_f32_16x16x32_f16(
                false, a.v, false, bf.v, (short)0, acc[j], false, false);
        }
    }

    // ---- bias + sigmoid, scatter to LDS gates[row][node] ----
    #pragma unroll
    for (int j = 0; j < 4; ++j) {
        const int N = nbase + j * 16 + (lane & 15);
        const float bv = (N < NI) ? bias[N] : 0.f;
        const int moff = (lane >> 4) * 8;
        #pragma unroll
        for (int r = 0; r < 8; ++r) {
            float z = acc[j][r] + bv;
            lgates[(moff + r) * NL + N] = 1.f / (1.f + __expf(-z));
        }
    }
    __syncthreads();

    // ---- per-row leaf path-product argmax; wave handles 4 rows ----
    for (int rr = 0; rr < 4; ++rr) {
        const int row = wv * 4 + rr;
        const float* g = &lgates[row * NL];
        float best = -1.f; int bi = 0;
        for (int l8 = 0; l8 < 8; ++l8) {
            const int leaf = lane * 8 + l8;
            float p = 1.f;
            #pragma unroll
            for (int d = 0; d < 8; ++d) {
                const int node = ((1 << d) - 1) + (leaf >> (8 - d));
                const float gg = g[node];
                p *= ((leaf >> (7 - d)) & 1) ? gg : (1.f - gg);
            }
            if (p > best) { best = p; bi = leaf; }   // ascending leaf => first-max kept
        }
        #pragma unroll
        for (int off = 16; off > 0; off >>= 1) {     // wave32 reduction
            float ob = __shfl_xor(best, off, 32);
            int   oi = __shfl_xor(bi,   off, 32);
            if (ob > best || (ob == best && oi < bi)) { best = ob; bi = oi; }
        }
        if (lane == 0) leafidx[row0 + row] = bi;
    }
}

// ---------------- gather: out[i] = dist[idx[i]] (bandwidth pass) ------------
__global__ __launch_bounds__(128) void gather_rows(const int* __restrict__ idx,
                                                   const float* __restrict__ dist,
                                                   float* __restrict__ out) {
    const int i  = blockIdx.x;
    const int li = idx[i];
    const float4* src = (const float4*)(dist + (size_t)li * NC);
    float4*       dst = (float4*)(out + (size_t)i * NC);
    for (int c = threadIdx.x; c < NC / 4; c += 128) dst[c] = src[c];
}

extern "C" void kernel_launch(void* const* d_in, const int* in_sizes, int n_in,
                              void* d_out, int out_size, void* d_ws, size_t ws_size,
                              hipStream_t stream) {
    const float* x     = (const float*)d_in[0];
    const float* W     = (const float*)d_in[1];
    const float* b     = (const float*)d_in[2];
    const float* cr    = (const float*)d_in[3];
    const float* gamma = (const float*)d_in[4];
    const float* beta  = (const float*)d_in[5];
    float* out = (float*)d_out;

    // workspace layout (all segments 16B aligned)
    float* psum  = (float*)d_ws;                       // 128*512 f  (256 KB)
    float* psq   = psum + RED_BLOCKS * DD;             // 128*512 f  (256 KB)
    float* scale = psq  + RED_BLOCKS * DD;             // 512 f
    float* shift = scale + DD;                         // 512 f
    _Float16* W16 = (_Float16*)(shift + DD);           // 256*512 h  (256 KB)
    int*   lidx  = (int*)(W16 + (size_t)NL * DD);      // 16384 i    (64 KB)
    float* dist  = (float*)(lidx + BS);                // 256*1000 f (~1 MB)

    colstat_partial <<<RED_BLOCKS, DD,        0, stream>>>(x, psum, psq);
    colstat_final   <<<2, 256,                0, stream>>>(psum, psq, gamma, beta, scale, shift);
    wconv           <<<(NL * DD) / 256, 256,  0, stream>>>(W, W16);
    softmax_rows    <<<NL, 256,               0, stream>>>(cr, dist);
    tree_gates_argmax<<<BS / 16, 128,         0, stream>>>(x, W16, b, scale, shift, lidx);
    gather_rows     <<<BS, 128,               0, stream>>>(lidx, dist, out);
}
